// LayerSymmetricLatentSelfAttention_53661321396532
// MI455X (gfx1250) — matrix operand order repaired and checked by reading, not compiled, not run on hardware
//
#include <hip/hip_runtime.h>
#include <hip/hip_bf16.h>

// ---------------------------------------------------------------------------
// CDNA5 (gfx1250) fused latent self-attention, bf16 WMMA, wave32.
// v2: packed bf16 LDS staging (u32 stores), double-buffered K/V tiles in attn.
// ---------------------------------------------------------------------------

typedef __attribute__((ext_vector_type(16))) __bf16 v16bf;
typedef __attribute__((ext_vector_type(8)))  float  v8f;

#define NH   12
#define RNK  64
#define HD   64
#define HID  768
#define NSEQ 1024
#define NB   8
#define MTOT (NB * NSEQ)   // 8192

static __device__ __forceinline__ unsigned f2bf_bits(float f) {
    union { float f; unsigned u; } x; x.f = f;
    return (x.u + 0x7fffu + ((x.u >> 16) & 1u)) >> 16;   // RNE
}
static __device__ __forceinline__ __bf16 f2bf(float f) {
    unsigned short hs = (unsigned short)f2bf_bits(f);
    return __builtin_bit_cast(__bf16, hs);
}
static __device__ __forceinline__ unsigned pack2bf(float lo, float hi) {
    return (f2bf_bits(lo) & 0xffffu) | (f2bf_bits(hi) << 16);
}

// A-fragment, bf16 16x32: lanes 0-15 row M=lane, K in {0..15};
// lanes 16-31 same rows, K in {16..31}. VGPR j holds 2 bf16.
static __device__ __forceinline__ v16bf load_a_frag(const __bf16* row0, int ld,
                                                    int k_base, int lane) {
    int m = lane & 15;
    int g = (lane >> 4) << 4;              // 0 or 16
    const __bf16* p = row0 + m * ld + k_base;
    v16bf a;
#pragma unroll
    for (int j = 0; j < 8; ++j) {
        int k0 = g + 2 * j;
        a[2 * j]     = p[k0];
        a[2 * j + 1] = p[k0 + 1];
    }
    return a;
}

// B-fragment, bf16 32x16, source stored [n][k] (i.e. we need B = src^T).
// lane = column n (mod 16), half-wave selects K 0..15 / 16..31 -> contiguous 16 bf16.
static __device__ __forceinline__ v16bf load_b_frag_t(const __bf16* src, int ld,
                                                      int n_base, int k_base, int lane) {
    int n = lane & 15;
    int g = (lane >> 4) << 4;              // 0 or 16
    const __bf16* p = src + (size_t)(n_base + n) * ld + k_base + g;
    return *(const v16bf*)p;               // 32B aligned (k_base+g mult. of 16 elems)
}

// B-fragment, bf16 32x16, source stored [k][n] (natural orientation).
static __device__ __forceinline__ v16bf load_b_frag_n(const __bf16* src, int ld,
                                                      int k_base, int n_base, int lane) {
    int n = lane & 15;
    int g = (lane >> 4) << 4;
    v16bf b;
#pragma unroll
    for (int j = 0; j < 8; ++j) {
        b[2 * j]     = src[(size_t)(k_base + g + 2 * j) * ld + n_base + n];
        b[2 * j + 1] = src[(size_t)(k_base + g + 2 * j + 1) * ld + n_base + n];
    }
    return b;
}

static __device__ __forceinline__ v8f wmma_bf16(v16bf a, v16bf b, v8f c) {
    return __builtin_amdgcn_wmma_f32_16x16x32_bf16(false, a, false, b,
                                                   (short)0, c, false, false);
}

static __device__ __forceinline__ float red_max16(float x) {
    x = fmaxf(x, __shfl_xor(x, 1, 16));
    x = fmaxf(x, __shfl_xor(x, 2, 16));
    x = fmaxf(x, __shfl_xor(x, 4, 16));
    x = fmaxf(x, __shfl_xor(x, 8, 16));
    return x;
}
static __device__ __forceinline__ float red_sum16(float x) {
    x += __shfl_xor(x, 1, 16);
    x += __shfl_xor(x, 2, 16);
    x += __shfl_xor(x, 4, 16);
    x += __shfl_xor(x, 8, 16);
    return x;
}

// ---------------------------------------------------------------------------
// Kernel 1: Y = X * W^T (+bias), bf16 output.
// mode 0: latent -> Y[m][64];  mode 1: value -> head-major Y[h][m][64].
// fp32 -> bf16 conversion done in registers, packed u32 DS stores.
// ---------------------------------------------------------------------------
__global__ __launch_bounds__(256) void proj_gemm_kernel(
    const float* __restrict__ X, const float* __restrict__ W,
    const float* __restrict__ bias, __bf16* __restrict__ Y, int K, int mode) {
    __shared__ __align__(32) __bf16 lds_x[128 * 32];
    __shared__ __align__(32) __bf16 lds_w[64 * 32];
    const int tid = threadIdx.x, lane = tid & 31, wave = tid >> 5;
    const int mbase = blockIdx.x * 128;
    const int nbase = blockIdx.y * 64;
    unsigned* lxw = (unsigned*)lds_x;
    unsigned* lww = (unsigned*)lds_w;

    v8f acc[4] = {};
    for (int kc = 0; kc < K; kc += 32) {
        // X tile: 128 rows x 32 cols -> 128*16 packed u32
#pragma unroll
        for (int j = 0; j < (128 * 16) / 256; ++j) {
            int i = tid + j * 256;
            int r = i >> 4, c2 = (i & 15) * 2;
            const float* xp = X + (size_t)(mbase + r) * K + kc + c2;
            float2 xv = *(const float2*)xp;
            lxw[i] = pack2bf(xv.x, xv.y);
        }
        // W tile: 64 rows x 32 cols -> 64*16 packed u32
#pragma unroll
        for (int j = 0; j < (64 * 16) / 256; ++j) {
            int i = tid + j * 256;
            int r = i >> 4, c2 = (i & 15) * 2;
            const float* wp = W + (size_t)(nbase + r) * K + kc + c2;
            float2 wv = *(const float2*)wp;
            lww[i] = pack2bf(wv.x, wv.y);
        }
        __syncthreads();
        v16bf a = load_a_frag(lds_x + wave * 16 * 32, 32, 0, lane);
#pragma unroll
        for (int t = 0; t < 4; ++t) {
            v16bf b = load_b_frag_t(lds_w, 32, t * 16, 0, lane);
            acc[t] = wmma_bf16(a, b, acc[t]);
        }
        __syncthreads();
    }

    const int hi = lane >> 4, n0 = lane & 15;
#pragma unroll
    for (int t = 0; t < 4; ++t) {
#pragma unroll
        for (int v = 0; v < 8; ++v) {
            int row = mbase + wave * 16 + v + hi * 8;
            int col = nbase + t * 16 + n0;
            float val = acc[t][v];
            if (mode == 1) {
                val += bias[col];
                int h = col >> 6, d = col & 63;
                Y[((size_t)h * MTOT + row) * 64 + d] = f2bf(val);
            } else {
                Y[(size_t)row * 64 + col] = f2bf(val);
            }
        }
    }
}

// ---------------------------------------------------------------------------
// Kernel 2: head_mats = core/NH + (resid - mean_h(resid)), bf16 [12][64][64]
// ---------------------------------------------------------------------------
__global__ __launch_bounds__(256) void headmats_kernel(
    const float* __restrict__ core, const float* __restrict__ resid,
    __bf16* __restrict__ hm) {
    int idx = blockIdx.x * 256 + threadIdx.x;   // (r,s) flat, 4096
    if (idx >= RNK * RNK) return;
    float mean = 0.f;
#pragma unroll
    for (int h = 0; h < NH; ++h) mean += resid[h * RNK * RNK + idx];
    mean *= (1.f / NH);
    float c = core[idx] * (1.f / NH);
#pragma unroll
    for (int h = 0; h < NH; ++h)
        hm[h * RNK * RNK + idx] = f2bf(c + resid[h * RNK * RNK + idx] - mean);
}

// ---------------------------------------------------------------------------
// Kernel 3: q[h][m][r] = 0.125 * latent[m][:] @ head_mat[h]   (1/sqrt(64) folded)
// ---------------------------------------------------------------------------
__global__ __launch_bounds__(256) void q_kernel(
    const __bf16* __restrict__ latent, const __bf16* __restrict__ hm,
    __bf16* __restrict__ q) {
    __shared__ __align__(32) __bf16 lds_l[128 * 64];
    __shared__ __align__(32) __bf16 lds_h[64 * 64];
    const int tid = threadIdx.x, lane = tid & 31, wave = tid >> 5;
    const int mbase = blockIdx.x * 128;
    const int h = blockIdx.y;

    {
        const uint2* s = (const uint2*)(latent + (size_t)mbase * 64);
        uint2* d = (uint2*)lds_l;
#pragma unroll
        for (int j = 0; j < (128 * 64 / 4) / 256; ++j) d[tid + j * 256] = s[tid + j * 256];
        const uint2* s2 = (const uint2*)(hm + (size_t)h * RNK * RNK);
        uint2* d2 = (uint2*)lds_h;
#pragma unroll
        for (int j = 0; j < (64 * 64 / 4) / 256; ++j) d2[tid + j * 256] = s2[tid + j * 256];
    }
    __syncthreads();

    v8f acc[4] = {};
#pragma unroll
    for (int kc = 0; kc < 2; ++kc) {
        v16bf a = load_a_frag(lds_l + wave * 16 * 64, 64, kc * 32, lane);
#pragma unroll
        for (int t = 0; t < 4; ++t) {
            v16bf b = load_b_frag_n(lds_h, 64, kc * 32, t * 16, lane);
            acc[t] = wmma_bf16(a, b, acc[t]);
        }
    }

    const int hi = lane >> 4, n0 = lane & 15;
#pragma unroll
    for (int t = 0; t < 4; ++t)
#pragma unroll
        for (int v = 0; v < 8; ++v) {
            int row = mbase + wave * 16 + v + hi * 8;
            q[((size_t)h * MTOT + row) * 64 + t * 16 + n0] = f2bf(acc[t][v] * 0.125f);
        }
}

// ---------------------------------------------------------------------------
// Kernel 4: flash attention, double-buffered K/V tiles.
// Block = 128 rows x one (b,h); 8 waves x 16 rows. Key tiles of 64.
// Next tile's global loads are issued into VGPRs before compute, stored to the
// alternate LDS buffer after PV -> global latency overlaps the 16 WMMAs/wave.
// ---------------------------------------------------------------------------
__global__ __launch_bounds__(256) void attn_kernel(
    const __bf16* __restrict__ q, const __bf16* __restrict__ latent,
    const __bf16* __restrict__ vhead, float* __restrict__ out) {
    __shared__ __align__(32) __bf16 lds_k[2][64 * 64];
    __shared__ __align__(32) __bf16 lds_v[2][64 * 64];
    __shared__ __align__(32) __bf16 lds_p[8 * 16 * 64];   // per-wave P staging
    const int tid = threadIdx.x, lane = tid & 31, wave = tid >> 5;
    const int mtile = blockIdx.x;   // 0..7
    const int h = blockIdx.y;       // 0..11
    const int b = blockIdx.z;       // 0..7
    const int mrow0 = b * NSEQ + mtile * 128;
    const int NT = NSEQ / 64;       // 16 key tiles

    const __bf16* kbase = latent + (size_t)b * NSEQ * 64;
    const __bf16* vbase = vhead + ((size_t)h * MTOT + b * NSEQ) * 64;

    // q fragments for this wave's 16 rows (K = 64 -> two 16x32 A frags)
    const __bf16* qptr = q + ((size_t)h * MTOT + mrow0 + wave * 16) * 64;
    const v16bf qa0 = load_a_frag(qptr, 64, 0, lane);
    const v16bf qa1 = load_a_frag(qptr, 64, 32, lane);

    float Mrow[8], Lrow[8];
    v8f oc[4] = {};
#pragma unroll
    for (int v = 0; v < 8; ++v) { Mrow[v] = -1e30f; Lrow[v] = 0.f; }

    const int hi = lane >> 4, n0 = lane & 15;
    __bf16* pw = lds_p + wave * 16 * 64;

    // prologue: stage tile 0 into buffer 0
    {
        const uint2* ks = (const uint2*)kbase;
        const uint2* vs = (const uint2*)vbase;
        uint2* kd = (uint2*)lds_k[0];
        uint2* vd = (uint2*)lds_v[0];
#pragma unroll
        for (int j = 0; j < (64 * 64 / 4) / 256; ++j) {
            kd[tid + j * 256] = ks[tid + j * 256];
            vd[tid + j * 256] = vs[tid + j * 256];
        }
    }
    __syncthreads();

    for (int kt = 0; kt < NT; ++kt) {
        const int cur = kt & 1;

        // issue next tile's global loads (kept in VGPRs through compute)
        uint2 kreg[4], vreg[4];
        if (kt + 1 < NT) {
            const uint2* ks = (const uint2*)(kbase + (size_t)(kt + 1) * 64 * 64);
            const uint2* vs = (const uint2*)(vbase + (size_t)(kt + 1) * 64 * 64);
#pragma unroll
            for (int j = 0; j < 4; ++j) {
                kreg[j] = ks[tid + j * 256];
                vreg[j] = vs[tid + j * 256];
            }
        }

        // S = q @ K^T  (16 x 64 per wave)
        v8f sc[4] = {};
#pragma unroll
        for (int t = 0; t < 4; ++t) {
            v16bf b0 = load_b_frag_t(lds_k[cur], 64, t * 16, 0, lane);
            sc[t] = wmma_bf16(qa0, b0, sc[t]);
            v16bf b1 = load_b_frag_t(lds_k[cur], 64, t * 16, 32, lane);
            sc[t] = wmma_bf16(qa1, b1, sc[t]);
        }

        // online softmax (rows live across 16-lane halves: row = v + 8*hi)
        float rmax[8];
#pragma unroll
        for (int v = 0; v < 8; ++v) {
            float m = sc[0][v];
#pragma unroll
            for (int t = 1; t < 4; ++t) m = fmaxf(m, sc[t][v]);
            rmax[v] = red_max16(m);
        }
#pragma unroll
        for (int v = 0; v < 8; ++v) {
            float nm = fmaxf(Mrow[v], rmax[v]);
            float sf = __expf(Mrow[v] - nm);
            Mrow[v] = nm;
            Lrow[v] *= sf;
#pragma unroll
            for (int t = 0; t < 4; ++t) oc[t][v] *= sf;
        }
        float rsum[8];
#pragma unroll
        for (int v = 0; v < 8; ++v) rsum[v] = 0.f;
#pragma unroll
        for (int t = 0; t < 4; ++t)
#pragma unroll
            for (int v = 0; v < 8; ++v) {
                float p = __expf(sc[t][v] - Mrow[v]);
                sc[t][v] = p;
                rsum[v] += p;
            }
#pragma unroll
        for (int v = 0; v < 8; ++v) Lrow[v] += red_sum16(rsum[v]);

        // stage P (C layout -> row-major per-wave [16][64]) for A-frag reload
#pragma unroll
        for (int t = 0; t < 4; ++t)
#pragma unroll
            for (int v = 0; v < 8; ++v)
                pw[(v + hi * 8) * 64 + t * 16 + n0] = f2bf(sc[t][v]);
        __syncthreads();

        // O += P @ V  (K = 64 key tile -> two 16x32 chunks)
        v16bf pa0 = load_a_frag(pw, 64, 0, lane);
        v16bf pa1 = load_a_frag(pw, 64, 32, lane);
#pragma unroll
        for (int t = 0; t < 4; ++t) {
            v16bf bv0 = load_b_frag_n(lds_v[cur], 64, 0, t * 16, lane);
            oc[t] = wmma_bf16(pa0, bv0, oc[t]);
            v16bf bv1 = load_b_frag_n(lds_v[cur], 64, 32, t * 16, lane);
            oc[t] = wmma_bf16(pa1, bv1, oc[t]);
        }

        // commit next tile to the alternate buffer
        if (kt + 1 < NT) {
            uint2* kd = (uint2*)lds_k[cur ^ 1];
            uint2* vd = (uint2*)lds_v[cur ^ 1];
#pragma unroll
            for (int j = 0; j < 4; ++j) {
                kd[tid + j * 256] = kreg[j];
                vd[tid + j * 256] = vreg[j];
            }
        }
        __syncthreads();
    }

    // epilogue: normalize, scatter fp32 to [b][n][768] at head column block
#pragma unroll
    for (int t = 0; t < 4; ++t)
#pragma unroll
        for (int v = 0; v < 8; ++v) {
            int row = mtile * 128 + wave * 16 + v + hi * 8;
            float val = oc[t][v] / Lrow[v];
            out[((size_t)b * NSEQ + row) * HID + h * 64 + t * 16 + n0] = val;
        }
}

// ---------------------------------------------------------------------------
extern "C" void kernel_launch(void* const* d_in, const int* in_sizes, int n_in,
                              void* d_out, int out_size, void* d_ws, size_t ws_size,
                              hipStream_t stream) {
    (void)in_sizes; (void)n_in; (void)out_size; (void)ws_size;
    const float* hidden  = (const float*)d_in[0];   // [8,1024,768]
    const float* basis_w = (const float*)d_in[1];   // [64,768]
    const float* core    = (const float*)d_in[2];   // [64,64]
    const float* resid   = (const float*)d_in[3];   // [12,64,64]
    const float* value_w = (const float*)d_in[4];   // [768,768]
    const float* value_b = (const float*)d_in[5];   // [768]
    float* out = (float*)d_out;                     // [8,1024,768]

    __bf16* latent_bf = (__bf16*)d_ws;                       // 8192*64
    __bf16* vhead     = latent_bf + (size_t)MTOT * 64;       // 12*8192*64
    __bf16* hm        = vhead + (size_t)NH * MTOT * 64;      // 12*64*64
    __bf16* qbuf      = hm + (size_t)NH * RNK * RNK;         // 12*8192*64

    // latent projection (bf16)
    proj_gemm_kernel<<<dim3(MTOT / 128, 1), 256, 0, stream>>>(
        hidden, basis_w, nullptr, latent_bf, HID, 0);
    // per-head core matrices
    headmats_kernel<<<dim3((RNK * RNK + 255) / 256), 256, 0, stream>>>(core, resid, hm);
    // value projection, head-major layout
    proj_gemm_kernel<<<dim3(MTOT / 128, HID / 64), 256, 0, stream>>>(
        hidden, value_w, value_b, vhead, HID, 1);
    // q = latent @ head_mat (scaled)
    q_kernel<<<dim3(MTOT / 128, NH), 256, 0, stream>>>(latent_bf, hm, qbuf);
    // fused flash attention
    attn_kernel<<<dim3(NSEQ / 128, NH, NB), 256, 0, stream>>>(qbuf, latent_bf, vhead, out);
}
